// EnhancedObj_52106543235165
// MI455X (gfx1250) — compile-verified, matrix operand order checked by reading, not run on hardware
//
#include <hip/hip_runtime.h>

// ---------------------------------------------------------------------------
// Types
// ---------------------------------------------------------------------------
typedef __attribute__((ext_vector_type(16))) __bf16     v16bf;
typedef __attribute__((ext_vector_type(8)))  float      v8f;
typedef __attribute__((ext_vector_type(4)))  unsigned   v4u;
typedef __attribute__((ext_vector_type(4)))  int        v4i;
typedef __attribute__((address_space(1))) v4i           as1_v4i;   // global
typedef __attribute__((address_space(3))) v4i           as3_v4i;   // LDS

// Problem constants
constexpr int BS   = 8;
constexpr int F    = 64;     // win_len
constexpr int OBJ  = 36;
constexpr int D    = 2048;
constexpr int NPB  = F * OBJ;        // 2304 objects per batch
constexpr int NV   = BS * F;         // 512 visual rows
constexpr int NO   = BS * NPB;       // 18432 object rows

constexpr int BK  = 32;   // == WMMA K for bf16
constexpr int PAD = 8;    // bf16 padding per LDS row -> stride 40 (conflict-free)
constexpr int LDT = BK + PAD;

enum { BL_NK = 0, BL_KN = 1 };
enum { EPI_BIAS_TANH_BF16 = 0, EPI_SCALE_TRANS_F32 = 1, EPI_PLAIN_F32 = 2 };

// CDNA5 has a native V_TANH_F32 transcendental; use it when the builtin exists.
__device__ __forceinline__ float fast_tanh(float x) {
#if __has_builtin(__builtin_amdgcn_tanhf)
    return __builtin_amdgcn_tanhf(x);
#else
    return tanhf(x);
#endif
}

// CDNA5 async global->LDS copies (ASYNCcnt-tracked), when the toolchain has them.
#if __has_builtin(__builtin_amdgcn_global_load_async_to_lds_b128) && \
    __has_builtin(__builtin_amdgcn_s_wait_asynccnt)
#define HAVE_ASYNC_LDS 1
#else
#define HAVE_ASYNC_LDS 0
#endif

#if HAVE_ASYNC_LDS
__device__ __forceinline__ void async_cp16(const void* g, void* l) {
    // generic->AS1 is identity; generic LDS pointer low 32 bits are the LDS offset
    __builtin_amdgcn_global_load_async_to_lds_b128(
        (as1_v4i*)(unsigned long long)g,
        (as3_v4i*)(unsigned int)(unsigned long long)l,
        0, 0);
}
#endif

// ---------------------------------------------------------------------------
// Tiled bf16 WMMA GEMM with double-buffered LDS software pipeline.
//   A: [M x K] bf16 row-major (lda)
//   B: BL_NK -> stored [N][K] (i.e. B^T row-major, ldb = K-stride)
//      BL_KN -> stored [K][N] row-major (ldb = N-stride), transposed into LDS
//   Epilogues: tanh(acc+bias) -> bf16 ; acc*scale -> fp32 transposed ; fp32
// ---------------------------------------------------------------------------
template<int BLAYOUT, int EPI, int BM, int BN, int WROWS, int WCOLS>
__global__ __launch_bounds__(256)
void wmma_gemm(const __bf16* __restrict__ A, const __bf16* __restrict__ B,
               void* __restrict__ Cout, const float* __restrict__ bias,
               int M, int N, int K, int lda, int ldb, int ldc, float scale,
               long long aBatch, long long bBatch, long long cBatch)
{
    constexpr int WM = BM / WROWS;
    constexpr int WN = BN / WCOLS;
    constexpr int TM = WM / 16;
    constexpr int TN = WN / 16;
    constexpr int A_IT = (BM * 4) / 256;                       // 16B chunks / thread
    constexpr int B_IT = (BLAYOUT == BL_NK) ? (BN * 4) / 256
                                            : (BK * BN / 8) / 256;

    __shared__ __bf16 lA[2][BM * LDT];
    __shared__ __bf16 lB[2][BN * LDT];

    const int bz = blockIdx.z;
    A += (long long)bz * aBatch;
    B += (long long)bz * bBatch;
    float*  Cf = (float*)Cout  + (long long)bz * cBatch;
    __bf16* Cb = (__bf16*)Cout + (long long)bz * cBatch;

    const int tid  = threadIdx.x;
    const int lane = tid & 31;
    const int h    = lane >> 4;   // half-wave select (K-phase)
    const int l16  = lane & 15;
    const int wave = tid >> 5;

    const int m0 = blockIdx.y * BM;
    const int n0 = blockIdx.x * BN;
    const int wM = (wave / WCOLS) * WM;
    const int wN = (wave % WCOLS) * WN;

    v8f acc[TM][TN] = {};
    v4u ra[A_IT], rb[B_IT];

    // ---- register-staged helpers (fallback path + KN transpose path) ----
    auto gloadA = [&](int k0) {
        #pragma unroll
        for (int i = 0; i < A_IT; ++i) {
            int lin = i * 256 + tid;
            int r = lin >> 2, c = (lin & 3) * 8;
            ra[i] = *reinterpret_cast<const v4u*>(A + (long long)(m0 + r) * lda + k0 + c);
        }
    };
    auto lstoreA = [&](int buf) {
        #pragma unroll
        for (int i = 0; i < A_IT; ++i) {
            int lin = i * 256 + tid;
            int r = lin >> 2, c = (lin & 3) * 8;
            *reinterpret_cast<v4u*>(&lA[buf][r * LDT + c]) = ra[i];
        }
    };
    auto gloadB = [&](int k0) {
        if (BLAYOUT == BL_NK) {
            #pragma unroll
            for (int i = 0; i < B_IT; ++i) {
                int lin = i * 256 + tid;
                int r = lin >> 2, c = (lin & 3) * 8;
                rb[i] = *reinterpret_cast<const v4u*>(B + (long long)(n0 + r) * ldb + k0 + c);
            }
        } else {
            constexpr int CPR = BN / 8;
            #pragma unroll
            for (int i = 0; i < B_IT; ++i) {
                int lin = i * 256 + tid;
                int kr = lin / CPR, nc = (lin % CPR) * 8;
                rb[i] = *reinterpret_cast<const v4u*>(B + (long long)(k0 + kr) * ldb + n0 + nc);
            }
        }
    };
    auto lstoreB = [&](int buf) {
        if (BLAYOUT == BL_NK) {
            #pragma unroll
            for (int i = 0; i < B_IT; ++i) {
                int lin = i * 256 + tid;
                int r = lin >> 2, c = (lin & 3) * 8;
                *reinterpret_cast<v4u*>(&lB[buf][r * LDT + c]) = rb[i];
            }
        } else {
            constexpr int CPR = BN / 8;
            #pragma unroll
            for (int i = 0; i < B_IT; ++i) {
                int lin = i * 256 + tid;
                int kr = lin / CPR, nc = (lin % CPR) * 8;
                union { v4u u; __bf16 e[8]; } d; d.u = rb[i];
                #pragma unroll
                for (int j = 0; j < 8; ++j)
                    lB[buf][(nc + j) * LDT + kr] = d.e[j];   // transpose scatter
            }
        }
    };

    // fragments + WMMA burst on one LDS buffer
    // (ISA 16-bit A layout: elems0-7 = K[h*8 + 0..7], elems8-15 = K[16 + h*8 + 0..7])
    auto compute = [&](int buf) {
        v16bf fa[TM], fb[TN];
        #pragma unroll
        for (int i = 0; i < TM; ++i) {
            int row = wM + i * 16 + l16;
            union { v16bf v; v4u u[2]; } t;
            t.u[0] = *reinterpret_cast<const v4u*>(&lA[buf][row * LDT + h * 8]);
            t.u[1] = *reinterpret_cast<const v4u*>(&lA[buf][row * LDT + 16 + h * 8]);
            fa[i] = t.v;
        }
        #pragma unroll
        for (int j = 0; j < TN; ++j) {
            int row = wN + j * 16 + l16;
            union { v16bf v; v4u u[2]; } t;
            t.u[0] = *reinterpret_cast<const v4u*>(&lB[buf][row * LDT + h * 8]);
            t.u[1] = *reinterpret_cast<const v4u*>(&lB[buf][row * LDT + 16 + h * 8]);
            fb[j] = t.v;
        }
        #pragma unroll
        for (int i = 0; i < TM; ++i)
            #pragma unroll
            for (int j = 0; j < TN; ++j)
                acc[i][j] = __builtin_amdgcn_wmma_f32_16x16x32_bf16(
                    false, fa[i], false, fb[j], (short)0, acc[i][j], false, false);
    };

    const int KT = K / BK;

#if HAVE_ASYNC_LDS
    // ---- async pipeline: tiles DMA straight into LDS, one barrier per K-tile
    auto stageA = [&](int k0, int buf) {
        #pragma unroll
        for (int i = 0; i < A_IT; ++i) {
            int lin = i * 256 + tid;
            int r = lin >> 2, c = (lin & 3) * 8;
            async_cp16(A + (long long)(m0 + r) * lda + k0 + c, &lA[buf][r * LDT + c]);
        }
    };
    auto stageB = [&](int k0, int buf) {
        #pragma unroll
        for (int i = 0; i < B_IT; ++i) {
            int lin = i * 256 + tid;
            int r = lin >> 2, c = (lin & 3) * 8;
            async_cp16(B + (long long)(n0 + r) * ldb + k0 + c, &lB[buf][r * LDT + c]);
        }
    };

    stageA(0, 0);
    if (BLAYOUT == BL_NK) stageB(0, 0);
    else { gloadB(0); lstoreB(0); }
    __builtin_amdgcn_s_wait_asynccnt(0);
    __syncthreads();
    for (int kt = 0; kt < KT; ++kt) {
        const int nb = (kt + 1) & 1;
        if (kt + 1 < KT) {
            stageA((kt + 1) * BK, nb);
            if (BLAYOUT == BL_NK) stageB((kt + 1) * BK, nb);
            else gloadB((kt + 1) * BK);
        }
        compute(kt & 1);
        if (kt + 1 < KT && BLAYOUT == BL_KN) lstoreB(nb);
        __builtin_amdgcn_s_wait_asynccnt(0);
        __syncthreads();
    }
#else
    // ---- register-staged double-buffered pipeline
    gloadA(0); gloadB(0);
    lstoreA(0); lstoreB(0);
    __syncthreads();
    for (int kt = 0; kt < KT; ++kt) {
        if (kt + 1 < KT) { gloadA((kt + 1) * BK); gloadB((kt + 1) * BK); }
        compute(kt & 1);
        if (kt + 1 < KT) { lstoreA((kt + 1) & 1); lstoreB((kt + 1) & 1); }
        __syncthreads();
    }
#endif

    // ---- epilogue (D layout: n = l16, m = base + r + 8h) ----
    #pragma unroll
    for (int i = 0; i < TM; ++i) {
        #pragma unroll
        for (int j = 0; j < TN; ++j) {
            int n  = n0 + wN + j * 16 + l16;
            int mb = m0 + wM + i * 16 + 8 * h;
            if (EPI == EPI_BIAS_TANH_BF16) {
                float bv = bias[n];
                #pragma unroll
                for (int r = 0; r < 8; ++r)
                    Cb[(long long)(mb + r) * ldc + n] = (__bf16)fast_tanh(acc[i][j][r] + bv);
            } else if (EPI == EPI_SCALE_TRANS_F32) {
                union { v8f v; float4 f4[2]; } u;
                #pragma unroll
                for (int r = 0; r < 8; ++r) u.v[r] = acc[i][j][r] * scale;
                float* p = Cf + (long long)n * ldc + mb;    // 16B-aligned
                *reinterpret_cast<float4*>(p)     = u.f4[0];
                *reinterpret_cast<float4*>(p + 4) = u.f4[1];
            } else {
                #pragma unroll
                for (int r = 0; r < 8; ++r)
                    Cf[(long long)(mb + r) * ldc + n] = acc[i][j][r];
            }
        }
    }
}

// ---------------------------------------------------------------------------
// fp32 -> bf16 convert, 8 elems/thread (all sizes are multiples of 2048)
// ---------------------------------------------------------------------------
__global__ __launch_bounds__(256)
void cvt_bf16(const float* __restrict__ in, __bf16* __restrict__ out, long long n)
{
    long long i = ((long long)blockIdx.x * 256 + threadIdx.x) * 8;
    if (i + 8 > n) return;
    union { v4u u; __bf16 e[8]; } o;
    #pragma unroll
    for (int j = 0; j < 8; ++j) o.e[j] = (__bf16)in[i + j];
    *reinterpret_cast<v4u*>(out + i) = o.u;
}

// ---------------------------------------------------------------------------
// fp32 [R][C] -> bf16 transposed [C][R], 32x32 LDS tiles (for weights)
// ---------------------------------------------------------------------------
__global__ __launch_bounds__(256)
void cvt_bf16_T(const float* __restrict__ in, __bf16* __restrict__ out,
                int rows, int cols)
{
    __shared__ float tile[32][33];
    const int r0 = blockIdx.y * 32, c0 = blockIdx.x * 32;
    const int tr = threadIdx.x >> 5, tc = threadIdx.x & 31;   // 8 x 32
    #pragma unroll
    for (int i = 0; i < 4; ++i)
        tile[tr + i * 8][tc] = in[(long long)(r0 + tr + i * 8) * cols + c0 + tc];
    __syncthreads();
    #pragma unroll
    for (int i = 0; i < 4; ++i)
        out[(long long)(c0 + tr + i * 8) * rows + r0 + tc] = (__bf16)tile[tc][tr + i * 8];
}

// ---------------------------------------------------------------------------
// Row LayerNorm over D=2048 (input bf16 = tanh output), fp32 and/or bf16 out
// ---------------------------------------------------------------------------
__global__ __launch_bounds__(256)
void ln_bf16(const __bf16* __restrict__ x, const float* __restrict__ g,
             const float* __restrict__ be, float* __restrict__ outF,
             __bf16* __restrict__ outB)
{
    __shared__ float s1[256], s2[256];
    long long row = blockIdx.x;
    const __bf16* xr = x + row * D;
    int t = threadIdx.x;
    float v[8], sum = 0.f, sq = 0.f;
    #pragma unroll
    for (int i = 0; i < 8; ++i) {
        float f = (float)xr[t + i * 256];
        v[i] = f; sum += f; sq += f * f;
    }
    s1[t] = sum; s2[t] = sq; __syncthreads();
    for (int s = 128; s > 0; s >>= 1) {
        if (t < s) { s1[t] += s1[t + s]; s2[t] += s2[t + s]; }
        __syncthreads();
    }
    float mu = s1[0] * (1.0f / D);
    float rs = rsqrtf(s2[0] * (1.0f / D) - mu * mu + 1e-5f);
    #pragma unroll
    for (int i = 0; i < 8; ++i) {
        int c = t + i * 256;
        float o = (v[i] - mu) * rs * g[c] + be[c];
        if (outF) outF[row * D + c] = o;
        if (outB) outB[row * D + c] = (__bf16)o;
    }
}

// ---------------------------------------------------------------------------
// Final: out = LN(tanh(agg + ve)) over D=2048, fp32 out
// ---------------------------------------------------------------------------
__global__ __launch_bounds__(256)
void ln_final(const float* __restrict__ agg, const float* __restrict__ ve,
              const float* __restrict__ g, const float* __restrict__ be,
              float* __restrict__ out)
{
    __shared__ float s1[256], s2[256];
    long long row = blockIdx.x;
    int t = threadIdx.x;
    float v[8], sum = 0.f, sq = 0.f;
    #pragma unroll
    for (int i = 0; i < 8; ++i) {
        long long idx = row * D + t + i * 256;
        float f = fast_tanh(agg[idx] + ve[idx]);
        v[i] = f; sum += f; sq += f * f;
    }
    s1[t] = sum; s2[t] = sq; __syncthreads();
    for (int s = 128; s > 0; s >>= 1) {
        if (t < s) { s1[t] += s1[t + s]; s2[t] += s2[t + s]; }
        __syncthreads();
    }
    float mu = s1[0] * (1.0f / D);
    float rs = rsqrtf(s2[0] * (1.0f / D) - mu * mu + 1e-5f);
    #pragma unroll
    for (int i = 0; i < 8; ++i) {
        int c = t + i * 256;
        out[row * D + c] = (v[i] - mu) * rs * g[c] + be[c];
    }
}

// ---------------------------------------------------------------------------
// Softmax over contiguous rows of adj^T (len NPB=2304 = 9*256), bf16 out (P^T)
// ---------------------------------------------------------------------------
__global__ __launch_bounds__(256)
void softmax_row(const float* __restrict__ in, __bf16* __restrict__ out)
{
    __shared__ float sr[256];
    long long row = blockIdx.x;
    const float* xr = in + row * (long long)NPB;
    int t = threadIdx.x;
    float v[9], mx = -3.4e38f;
    #pragma unroll
    for (int i = 0; i < 9; ++i) { v[i] = xr[t + i * 256]; mx = fmaxf(mx, v[i]); }
    sr[t] = mx; __syncthreads();
    for (int s = 128; s > 0; s >>= 1) {
        if (t < s) sr[t] = fmaxf(sr[t], sr[t + s]);
        __syncthreads();
    }
    mx = sr[0]; __syncthreads();
    float sum = 0.f;
    #pragma unroll
    for (int i = 0; i < 9; ++i) { v[i] = __expf(v[i] - mx); sum += v[i]; }
    sr[t] = sum; __syncthreads();
    for (int s = 128; s > 0; s >>= 1) {
        if (t < s) sr[t] += sr[t + s];
        __syncthreads();
    }
    float inv = 1.0f / sr[0];
    #pragma unroll
    for (int i = 0; i < 9; ++i)
        out[row * (long long)NPB + t + i * 256] = (__bf16)(v[i] * inv);
}

// ---------------------------------------------------------------------------
// Host launcher
// ---------------------------------------------------------------------------
extern "C" void kernel_launch(void* const* d_in, const int* in_sizes, int n_in,
                              void* d_out, int out_size, void* d_ws, size_t ws_size,
                              hipStream_t stream)
{
    const float* visual = (const float*)d_in[0];
    const float* obj    = (const float*)d_in[1];
    const float* Wv     = (const float*)d_in[2];
    const float* bv     = (const float*)d_in[3];
    const float* Wo     = (const float*)d_in[4];
    const float* bo     = (const float*)d_in[5];
    const float* ln_v_g = (const float*)d_in[6];
    const float* ln_v_b = (const float*)d_in[7];
    const float* ln_o_g = (const float*)d_in[8];
    const float* ln_o_b = (const float*)d_in[9];
    const float* ln_ov_g= (const float*)d_in[10];
    const float* ln_ov_b= (const float*)d_in[11];

    // workspace layout (all 256B aligned by construction)
    char* ws = (char*)d_ws;
    size_t o = 0;
    auto take = [&](size_t bytes) { char* p = ws + o; o += (bytes + 255) & ~(size_t)255; return p; };
    __bf16* vis_b   = (__bf16*)take((size_t)NV * D * 2);     // visual bf16
    __bf16* obj_b   = (__bf16*)take((size_t)NO * D * 2);     // obj bf16; later reused as oe
    __bf16* WvT_b   = (__bf16*)take((size_t)D * D * 2);      // Wv^T bf16  [N][K]
    __bf16* WoT_b   = (__bf16*)take((size_t)D * D * 2);      // Wo^T bf16  [N][K]
    __bf16* tanhv_b = (__bf16*)take((size_t)NV * D * 2);     // tanh(visual@Wv+b)
    __bf16* tanho_b = (__bf16*)take((size_t)NO * D * 2);     // tanh(obj@Wo+b)
    float*  ve_f    = (float*) take((size_t)NV * D * 4);     // LN'd ve fp32
    __bf16* ve_b    = (__bf16*)take((size_t)NV * D * 2);     // LN'd ve bf16
    float*  adjT    = (float*) take((size_t)BS * F * NPB * 4);
    __bf16* PT      = (__bf16*)take((size_t)BS * F * NPB * 2);
    float*  agg     = (float*) take((size_t)BS * F * D * 4);
    __bf16* oe_b    = obj_b;   // alias: obj bf16 dead once GEMM1' finishes

    const float invsqrt_d = 0.022097086912079608f; // 1/sqrt(2048)

    // 1) fp32 -> bf16 converts (weights transposed to [N][K] for async NK path)
    cvt_bf16<<<(NV * D) / 2048, 256, 0, stream>>>(visual, vis_b, (long long)NV * D);
    cvt_bf16<<<(NO * D) / 2048, 256, 0, stream>>>(obj,    obj_b, (long long)NO * D);
    cvt_bf16_T<<<dim3(D / 32, D / 32), 256, 0, stream>>>(Wv, WvT_b, D, D);
    cvt_bf16_T<<<dim3(D / 32, D / 32), 256, 0, stream>>>(Wo, WoT_b, D, D);

    // 2) visual branch GEMM: tanh(vis @ Wv + bv) -> bf16   [512 x 2048]
    wmma_gemm<BL_NK, EPI_BIAS_TANH_BF16, 128, 128, 4, 2>
        <<<dim3(D / 128, NV / 128, 1), 256, 0, stream>>>(
        vis_b, WvT_b, tanhv_b, bv, NV, D, D, D, D, D, 1.0f, 0, 0, 0);

    // 3) object branch GEMM: tanh(obj @ Wo + bo) -> bf16   [18432 x 2048]
    wmma_gemm<BL_NK, EPI_BIAS_TANH_BF16, 128, 128, 4, 2>
        <<<dim3(D / 128, NO / 128, 1), 256, 0, stream>>>(
        obj_b, WoT_b, tanho_b, bo, NO, D, D, D, D, D, 1.0f, 0, 0, 0);

    // 4) LayerNorms
    ln_bf16<<<NV, 256, 0, stream>>>(tanhv_b, ln_v_g, ln_v_b, ve_f, ve_b);
    ln_bf16<<<NO, 256, 0, stream>>>(tanho_b, ln_o_g, ln_o_b, nullptr, oe_b);

    // 5) adjacency: adj^T[b][f][n] = (oe[b] @ ve[b]^T)^T / sqrt(d)
    //    B = ve bf16 row-major == [N][K] layout directly; transposed fp32 store
    wmma_gemm<BL_NK, EPI_SCALE_TRANS_F32, 128, 64, 4, 2>
        <<<dim3(1, NPB / 128, BS), 256, 0, stream>>>(
        oe_b, ve_b, adjT, nullptr, NPB, F, D, D, D, /*ldcT=*/NPB, invsqrt_d,
        (long long)NPB * D, (long long)F * D, (long long)F * NPB);

    // 6) softmax over n (contiguous rows of adj^T) -> P^T bf16
    softmax_row<<<BS * F, 256, 0, stream>>>(adjT, PT);

    // 7) aggregate: agg[b] = P^T[b] (64x2304) @ oe[b] (2304x2048), fp32 out
    wmma_gemm<BL_KN, EPI_PLAIN_F32, 64, 128, 2, 4>
        <<<dim3(D / 128, 1, BS), 256, 0, stream>>>(
        PT, oe_b, agg, nullptr, F, D, NPB, NPB, D, D, 1.0f,
        (long long)F * NPB, (long long)NPB * D, (long long)F * D);

    // 8) out = LN(tanh(agg + ve))
    ln_final<<<BS * F, 256, 0, stream>>>(agg, ve_f, ln_ov_g, ln_ov_b, (float*)d_out);
}
